// GenerativeModel_42485816492133
// MI455X (gfx1250) — matrix operand order repaired and checked by reading, not compiled, not run on hardware
//
#include <hip/hip_runtime.h>
#include <hip/hip_bf16.h>

// ---------------------------------------------------------------------------
// GenerativeModel pipeline for MI455X (gfx1250, wave32, WMMA).
// N=20000 B=100 NPG=200 L=31 T=32 H=128 E=320000 VN=100 VA=50.
//
// Restructurings:
//  * layer-0 GCN GEMM folded into the embedding table: xw0 = (emb@W0)[types]
//  * concats@linN_W split: logits[n,t]=relu(A[bs[n],t]+C[n]+b)·wf
//    with A=ys@W1, C=node_emb@W2  -> removes the 42-GFLOP GEMM.
//  * all dense GEMM weights pre-transposed + pre-converted to f16 once, so
//    WMMA fragment loads are contiguous 16-byte vector loads; K=128 is
//    compile-time so the k-loop unrolls into 4 v_wmma per tile.
// ---------------------------------------------------------------------------

#define HDIM 128           // hidden dim (fixed by the model); K of every GEMM

typedef _Float16 v16h __attribute__((ext_vector_type(16)));
typedef _Float16 h8   __attribute__((ext_vector_type(8)));
typedef float    v8f  __attribute__((ext_vector_type(8)));
typedef float    f4   __attribute__((ext_vector_type(4)));

// ---------------- degree / norm precompute ---------------------------------
__global__ void k_fill1(float* p, int n) {
  int i = blockIdx.x * blockDim.x + threadIdx.x;
  if (i < n) p[i] = 1.0f;                       // self-loop weight
}
__global__ void k_deg(const int* dst, const float* w, float* deg, int E) {
  int e = blockIdx.x * blockDim.x + threadIdx.x;
  if (e < E) atomicAdd(&deg[dst[e]], w[e]);
}
__global__ void k_dinv(float* d, int n) {
  int i = blockIdx.x * blockDim.x + threadIdx.x;
  if (i < n) { float v = d[i]; d[i] = (v > 0.f) ? rsqrtf(v) : 0.f; }
}
__global__ void k_norm(const int* src, const int* dst, const float* w,
                       const float* dinv, float* norm, int E) {
  int e = blockIdx.x * blockDim.x + threadIdx.x;
  if (e < E) norm[e] = dinv[src[e]] * w[e] * dinv[dst[e]];
}

// ---------------- weight prep: transpose f32[K,Nn] -> f16[Nn,K] ------------
__global__ void k_prep_w(const float* __restrict__ W, _Float16* __restrict__ Wt,
                         int K, int Nn) {
  int idx = blockIdx.x * blockDim.x + threadIdx.x;
  if (idx >= K * Nn) return;
  int n = idx / K, k = idx % K;                 // Wt row-contiguous in k
  Wt[idx] = (_Float16)W[(long long)k * Nn + n];
}

// ---------------- tiny GEMM (M=VN=100, not /16) : EW0 = emb @ W0 -----------
__global__ void k_small_gemm(const float* __restrict__ A, const float* __restrict__ B,
                             float* __restrict__ C, int K, int Nn) {
  int m = blockIdx.x, j = threadIdx.x;          // Nn threads
  float s = 0.f;
  for (int k = 0; k < K; ++k) s += A[m * K + k] * B[k * Nn + j];
  C[m * Nn + j] = s;
}

__global__ void k_gather(const float* __restrict__ EW0, const int* __restrict__ types,
                         float* __restrict__ xw) {
  int n = blockIdx.x, h = threadIdx.x;          // HDIM threads
  xw[(long long)n * HDIM + h] = EW0[types[n] * HDIM + h];
}

// ---------------- GCN aggregation ------------------------------------------
__global__ void k_agg_self(const float* __restrict__ xw, const float* __restrict__ dinv,
                           float* __restrict__ acc, long long total) {
  long long i = (long long)blockIdx.x * blockDim.x + threadIdx.x;
  if (i >= total) return;
  int n = (int)(i >> 7);                        // /HDIM
  float d = dinv[n];
  acc[i] = d * d * xw[i];
}
__global__ void k_agg_edges(const int* __restrict__ src, const int* __restrict__ dst,
                            const float* __restrict__ norm, const float* __restrict__ xw,
                            float* __restrict__ acc, long long total) {
  long long i = (long long)blockIdx.x * blockDim.x + threadIdx.x;
  if (i >= total) return;
  int e = (int)(i >> 7);
  int h = (int)(i & (HDIM - 1));
  atomicAdd(&acc[(long long)dst[e] * HDIM + h], norm[e] * xw[(long long)src[e] * HDIM + h]);
}
__global__ void k_bias_relu(float* __restrict__ buf, const float* __restrict__ bias,
                            long long total) {
  long long i = (long long)blockIdx.x * blockDim.x + threadIdx.x;
  if (i >= total) return;
  buf[i] = fmaxf(buf[i] + bias[i & (HDIM - 1)], 0.f);
}

// ---------------- WMMA GEMM, K = HDIM = 128 (compile time) -----------------
// C[M,Nn] = A[M,128] @ Bt^T (+bias)(+relu).  A row-major f32; Bt is the
// weight pre-transposed to f16 [Nn,128] so both operands load contiguously.
// One wave per 16x16 tile; tile index is wave-uniform so EXEC stays all-1s
// around the WMMA (ISA 7.12).  Fragment layouts per ISA 7.12.2:
//   A 16x32 f16: lanes 0-15 hold K {0..7,16..23}, lanes 16-31 {8..15,24..31}
//   B 32x16 f16: same K order as A (lanes 0-15 {0..7,16..23}, 16-31 {8..15,24..31})
__global__ void k_wmma_gemm(const float* __restrict__ A, const _Float16* __restrict__ Bt,
                            const float* __restrict__ bias, float* __restrict__ C,
                            int M, int Nn, int relu) {
  const int lane = threadIdx.x & 31;
  const int wave = threadIdx.x >> 5;
  const int tilesN = Nn >> 4;
  const int tile = blockIdx.x * (blockDim.x >> 5) + wave;
  if (tile >= (M >> 4) * tilesN) return;        // wave-uniform exit
  const int mt = tile / tilesN, nt = tile % tilesN;
  const int g = lane >> 4;                      // lane half-group
  const int l15 = lane & 15;
  const float*    ap = A  + (long long)(mt * 16 + l15) * HDIM;  // this lane's A row
  const _Float16* bp = Bt + (long long)(nt * 16 + l15) * HDIM;  // this lane's B col
  const float bv = bias ? bias[nt * 16 + l15] : 0.f;

  v8f acc = {};
#pragma unroll
  for (int k0 = 0; k0 < HDIM; k0 += 32) {
    // A: two contiguous 8-float runs per lane -> four b128 loads + pk converts
    f4 a0 = *(const f4*)(ap + k0 + g * 8);
    f4 a1 = *(const f4*)(ap + k0 + g * 8 + 4);
    f4 a2 = *(const f4*)(ap + k0 + 16 + g * 8);
    f4 a3 = *(const f4*)(ap + k0 + 16 + g * 8 + 4);
    v16h a;
#pragma unroll
    for (int i = 0; i < 4; ++i) {
      a[i]      = (_Float16)a0[i];
      a[4 + i]  = (_Float16)a1[i];
      a[8 + i]  = (_Float16)a2[i];
      a[12 + i] = (_Float16)a3[i];
    }
    // B: two contiguous 8-half runs per lane -> two b128 loads
    h8 b0 = *(const h8*)(bp + k0 + g * 8);
    h8 b1 = *(const h8*)(bp + k0 + 16 + g * 8);
    v16h b;
#pragma unroll
    for (int i = 0; i < 8; ++i) { b[i] = b0[i]; b[8 + i] = b1[i]; }
    acc = __builtin_amdgcn_wmma_f32_16x16x32_f16(false, a, false, b,
                                                 (short)0, acc, false, false);
  }
#pragma unroll
  for (int r = 0; r < 8; ++r) {                 // C/D: vgpr r -> row g*8+r
    int row = mt * 16 + g * 8 + r;
    float v = acc[r] + bv;
    v = relu ? fmaxf(v, 0.f) : v;
    C[(long long)row * Nn + nt * 16 + l15] = v;
  }
}

// ---------------- mean pool (bs is sorted blocks of NPG nodes) -------------
__global__ void k_pool(const float* __restrict__ ne, float* __restrict__ hG, int NPG) {
  int b = blockIdx.x, h = threadIdx.x;          // HDIM threads
  const float* p = ne + (long long)b * NPG * HDIM + h;
  float s = 0.f;
  for (int i = 0; i < NPG; ++i) s += p[(long long)i * HDIM];
  hG[b * HDIM + h] = s / (float)NPG;
}

// ---------------- action head ----------------------------------------------
__global__ void k_action(const float* __restrict__ hG, const float* __restrict__ WA,
                         const float* __restrict__ bA, const float* __restrict__ WAf,
                         const float* __restrict__ bAf, float* __restrict__ out, int VA) {
  __shared__ float hsh[HDIM], t1[HDIM];
  int b = blockIdx.x, j = threadIdx.x;          // HDIM threads
  hsh[j] = hG[b * HDIM + j];
  __syncthreads();
  float a = bA[j];
  for (int k = 0; k < HDIM; ++k) a += hsh[k] * WA[k * HDIM + j];
  t1[j] = fmaxf(a, 0.f);
  __syncthreads();
  if (j < VA) {
    float a2 = bAf[j];
    for (int k = 0; k < HDIM; ++k) a2 += t1[k] * WAf[k * VA + j];
    out[b * VA + j] = a2;
  }
}

// ---------------- GRU input: [SOS, per-graph seq-weighted node sums] -------
__global__ void k_build_gru_in(const float* __restrict__ ne, const float* __restrict__ seq,
                               const float* __restrict__ embA, const int* __restrict__ act,
                               float* __restrict__ ing, int NPG, int L, int T) {
  int bt = blockIdx.x, b = bt / T, t = bt % T, h = threadIdx.x;   // HDIM threads
  float v;
  if (t == 0) {
    v = embA[act[b] * HDIM + h];
  } else {
    int l = t - 1, base = b * NPG;
    v = 0.f;
    for (int i = 0; i < NPG; ++i) {
      float s = seq[(long long)(base + i) * L + l];   // block-uniform -> cheap skip
      if (s != 0.f) v += s * ne[(long long)(base + i) * HDIM + h];
    }
  }
  ing[((long long)b * T + t) * HDIM + h] = v;
}

// ---------------- GRU scan: one workgroup per batch row --------------------
__global__ void k_gru(const float* __restrict__ gx, const float* __restrict__ hG,
                      const float* __restrict__ Whh, const float* __restrict__ bhh,
                      const int* __restrict__ len_seq, float* __restrict__ ys, int T) {
  __shared__ float h[HDIM];
  __shared__ float gh[3 * HDIM];
  const int b = blockIdx.x, j = threadIdx.x;    // 3*HDIM = 384 threads
  if (j < HDIM) h[j] = hG[b * HDIM + j];
  __syncthreads();
  const int len = len_seq[b];
  for (int t = 0; t < T; ++t) {
    float acc = bhh[j];
#pragma unroll 8
    for (int k = 0; k < HDIM; ++k) acc += h[k] * Whh[k * 3 * HDIM + j];
    gh[j] = acc;
    __syncthreads();
    if (j < HDIM) {
      const float* gxt = gx + ((long long)b * T + t) * 3 * HDIM;
      float r  = 1.f / (1.f + __expf(-(gxt[j] + gh[j])));
      float z  = 1.f / (1.f + __expf(-(gxt[HDIM + j] + gh[HDIM + j])));
      float nn = tanhf(gxt[2 * HDIM + j] + r * gh[2 * HDIM + j]);
      float hn = (1.f - z) * nn + z * h[j];
      ys[((long long)b * T + t) * HDIM + j] = (t < len) ? hn : 0.f;  // pad mask
      h[j] = hn;
    }
    __syncthreads();
  }
}

// ---------------- logits[n,t] = relu(A[bs[n],t]+C[n]+b) . wf + bf ----------
__global__ void k_logits(const float* __restrict__ Amat, const float* __restrict__ Cc,
                         const float* __restrict__ lnb, const float* __restrict__ wf,
                         const float* __restrict__ bf, const int* __restrict__ bs,
                         float* __restrict__ logits, int T) {
  __shared__ float red[4];
  const int n = blockIdx.x, h = threadIdx.x;    // HDIM = 128 threads (4 waves)
  const int b = bs[n];
  const float c = Cc[(long long)n * HDIM + h] + lnb[h];
  const float w = wf[h];
  for (int t = 0; t < T; ++t) {
    float v = fmaxf(Amat[((long long)b * T + t) * HDIM + h] + c, 0.f) * w;
#pragma unroll
    for (int off = 16; off > 0; off >>= 1) v += __shfl_down(v, off);
    if ((h & 31) == 0) red[h >> 5] = v;
    __syncthreads();
    if (h == 0) logits[(long long)n * T + t] = red[0] + red[1] + red[2] + red[3] + bf[0];
    __syncthreads();
  }
}

// ---------------- per-(graph,t) softmax over NPG nodes ---------------------
__global__ void k_softmax(const float* __restrict__ logits, float* __restrict__ out,
                          int T, int NPG) {
  __shared__ float vals[256];                   // NPG = N/B = 200 <= 256
  __shared__ float red[64];
  const int b = blockIdx.x / T, t = blockIdx.x % T, tid = threadIdx.x; // 64 thr
  const int base = b * NPG;
  float m = -3.4e38f;
  for (int i = tid; i < NPG; i += 64) {
    float v = logits[(long long)(base + i) * T + t];
    vals[i] = v; m = fmaxf(m, v);
  }
  red[tid] = m; __syncthreads();
  for (int s = 32; s > 0; s >>= 1) { if (tid < s) red[tid] = fmaxf(red[tid], red[tid + s]); __syncthreads(); }
  m = red[0]; __syncthreads();
  float sum = 0.f;
  for (int i = tid; i < NPG; i += 64) sum += __expf(vals[i] - m);
  red[tid] = sum; __syncthreads();
  for (int s = 32; s > 0; s >>= 1) { if (tid < s) red[tid] += red[tid + s]; __syncthreads(); }
  const float inv = 1.f / red[0];
  for (int i = tid; i < NPG; i += 64)
    out[(long long)(base + i) * T + t] = __expf(vals[i] - m) * inv;
}

// ---------------------------------------------------------------------------
static void launch_wmma(const float* A, const _Float16* Bt, const float* bias, float* C,
                        int M, int Nn, int relu, hipStream_t stream) {
  int tiles = (M >> 4) * (Nn >> 4);
  int blocks = (tiles + 7) / 8;                 // 8 waves (256 threads) / block
  k_wmma_gemm<<<blocks, 256, 0, stream>>>(A, Bt, bias, C, M, Nn, relu);
}
static void launch_prep(const float* W, _Float16* Wt, int K, int Nn, hipStream_t stream) {
  k_prep_w<<<(K * Nn + 255) / 256, 256, 0, stream>>>(W, Wt, K, Nn);
}

extern "C" void kernel_launch(void* const* d_in, const int* in_sizes, int n_in,
                              void* d_out, int out_size, void* d_ws, size_t ws_size,
                              hipStream_t stream) {
  (void)n_in; (void)out_size; (void)ws_size;
  const int*   nodeTypes = (const int*)d_in[0];
  const int*   e_src     = (const int*)d_in[1];
  const float* edge_attr = (const float*)d_in[2];
  const int*   bs        = (const int*)d_in[3];
  const float* seq       = (const float*)d_in[4];
  const int*   len_seq   = (const int*)d_in[6];
  const int*   act_in    = (const int*)d_in[7];
  const float* embN      = (const float*)d_in[8];
  const float* embA      = (const float*)d_in[9];
  const float* gcnW[3]   = {(const float*)d_in[10], (const float*)d_in[12], (const float*)d_in[14]};
  const float* gcnB[3]   = {(const float*)d_in[11], (const float*)d_in[13], (const float*)d_in[15]};
  const float* Wih = (const float*)d_in[16];
  const float* Whh = (const float*)d_in[17];
  const float* bih = (const float*)d_in[18];
  const float* bhh = (const float*)d_in[19];
  const float* WA  = (const float*)d_in[20];
  const float* bA  = (const float*)d_in[21];
  const float* WAf = (const float*)d_in[22];
  const float* bAf = (const float*)d_in[23];
  const float* WN  = (const float*)d_in[24];
  const float* bN  = (const float*)d_in[25];
  const float* WNf = (const float*)d_in[26];
  const float* bNf = (const float*)d_in[27];

  const int N   = in_sizes[0];
  const int E   = in_sizes[2];
  const int Bg  = in_sizes[5];
  const int L   = in_sizes[4] / N;
  const int T   = L + 1;
  const int H   = HDIM;
  const int VN  = in_sizes[8] / H;
  const int VA  = in_sizes[23];
  const int NPG = N / Bg;
  const int* e_dst = e_src + E;                 // edge_index is [2,E]

  // workspace carve-out: f32 region first, then f16 weight region
  float* ws     = (float*)d_ws;
  float* dinv   = ws; ws += N;                  // doubles as deg
  float* norm   = ws; ws += E;
  float* EW0    = ws; ws += (size_t)VN * H;
  float* buf0   = ws; ws += (size_t)N * H;      // xw / Cc
  float* buf1   = ws; ws += (size_t)N * H;      // activations / node_emb
  float* hG     = ws; ws += (size_t)Bg * H;
  float* ingru  = ws; ws += (size_t)Bg * T * H;
  float* gx     = ws; ws += (size_t)Bg * T * 3 * H;
  float* ys     = ws; ws += (size_t)Bg * T * H;
  float* Amat   = ws; ws += (size_t)Bg * T * H;
  float* logits = ws; ws += (size_t)N * T;
  _Float16* hw  = (_Float16*)ws;                // 16B-aligned (all counts %4==0)
  _Float16* W1t   = hw; hw += (size_t)H * H;    // gcn layer-1 weight, f16 [N,K]
  _Float16* W2t   = hw; hw += (size_t)H * H;    // gcn layer-2 weight
  _Float16* Wiht  = hw; hw += (size_t)H * 3 * H;
  _Float16* WN1t  = hw; hw += (size_t)H * H;    // linN_W rows 0..H-1
  _Float16* WN2t  = hw; hw += (size_t)H * H;    // linN_W rows H..2H-1

  float* actionOut = (float*)d_out;             // [B,VA]
  float* nodesOut  = actionOut + (size_t)Bg * VA; // [N,T]

  const long long NH = (long long)N * H;
  const long long EH = (long long)E * H;
  const int TPB = 256;

  // --- weight prep (transpose + f16) ---
  launch_prep(gcnW[1], W1t, H, H, stream);
  launch_prep(gcnW[2], W2t, H, H, stream);
  launch_prep(Wih, Wiht, H, 3 * H, stream);
  launch_prep(WN, WN1t, H, H, stream);
  launch_prep(WN + (size_t)H * H, WN2t, H, H, stream);

  // --- GCN normalization terms ---
  k_fill1<<<(N + TPB - 1) / TPB, TPB, 0, stream>>>(dinv, N);
  k_deg  <<<(E + TPB - 1) / TPB, TPB, 0, stream>>>(e_dst, edge_attr, dinv, E);
  k_dinv <<<(N + TPB - 1) / TPB, TPB, 0, stream>>>(dinv, N);
  k_norm <<<(E + TPB - 1) / TPB, TPB, 0, stream>>>(e_src, e_dst, edge_attr, dinv, norm, E);

  // --- layer 0 GEMM folded into embedding table ---
  k_small_gemm<<<VN, H, 0, stream>>>(embN, gcnW[0], EW0, H, H);
  k_gather<<<N, H, 0, stream>>>(EW0, nodeTypes, buf0);

  // --- three GCN layers: xw(buf0) -> aggregate -> relu (buf1) ---
  for (int l = 0; l < 3; ++l) {
    if (l == 1) launch_wmma(buf1, W1t, nullptr, buf0, N, H, 0, stream);
    if (l == 2) launch_wmma(buf1, W2t, nullptr, buf0, N, H, 0, stream);
    k_agg_self <<<(int)((NH + TPB - 1) / TPB), TPB, 0, stream>>>(buf0, dinv, buf1, NH);
    k_agg_edges<<<(int)((EH + TPB - 1) / TPB), TPB, 0, stream>>>(e_src, e_dst, norm, buf0, buf1, EH);
    k_bias_relu<<<(int)((NH + TPB - 1) / TPB), TPB, 0, stream>>>(buf1, gcnB[l], NH);
  }
  // node_emb now in buf1

  // --- pooled graph state + action head ---
  k_pool  <<<Bg, H, 0, stream>>>(buf1, hG, NPG);
  k_action<<<Bg, H, 0, stream>>>(hG, WA, bA, WAf, bAf, actionOut, VA);

  // --- GRU ---
  k_build_gru_in<<<Bg * T, H, 0, stream>>>(buf1, seq, embA, act_in, ingru, NPG, L, T);
  launch_wmma(ingru, Wiht, bih, gx, Bg * T, 3 * H, 0, stream);   // gx = in@Wih+bih
  k_gru<<<Bg, 3 * H, 0, stream>>>(gx, hG, Whh, bhh, len_seq, ys, T);

  // --- factored node scorer: A = ys@W1, C = node_emb@W2 ---
  launch_wmma(ys,   WN1t, nullptr, Amat, Bg * T, H, 0, stream);
  launch_wmma(buf1, WN2t, nullptr, buf0, N,      H, 0, stream);
  k_logits<<<N, H, 0, stream>>>(Amat, buf0, bN, WNf, bNf, bs, logits, T);

  // --- per-graph softmax over nodes ---
  k_softmax<<<Bg * T, 64, 0, stream>>>(logits, nodesOut, T, NPG);
}
